// lpLSTM_71949292143074
// MI455X (gfx1250) — matrix operand order repaired and checked by reading, not compiled, not run on hardware
//
#include <hip/hip_runtime.h>

// ---------------------------------------------------------------------------
// Low-pass LSTM for MI455X (gfx1250, wave32, WMMA).
//
//   Phase 1 (parallel): x_gates[S*B,4H] = bf16(input) @ bf16(W_ih^T) + bias
//                       -> one large v_wmma_f32_16x16x32_bf16 GEMM
//                          (4x4 register tiles, 16 wmma / K-step / wave).
//   Phase 2 (sequential, 512 steps, ONE fused kernel per step):
//       Workgroup owns column-tile jt (16 h-columns) across all 4 gate
//       sections and all 64 batch rows. K=1024 is split 4-ways across the
//       4 waves (8 k-tiles each -> 4x shorter dependent WMMA chain on the
//       latency-critical recurrence), partials reduced through LDS, then
//       each wave runs the in-register LSTM nonlinearity + retention filter
//       for one M-tile. h is re-packed to the CDNA5 A-fragment layout for
//       the next step (ping-pong buffers).
//
// All matrix fragments are pre-swizzled into the ISA per-lane layouts so the
// GEMM inner loops are pure contiguous 32B/lane loads + WMMA.
// ---------------------------------------------------------------------------

#define S_LEN 512
#define B_DIM 64
#define I_DIM 1024
#define H_DIM 1024
#define G_DIM 4096                  // 4*H
#define M_BIG (S_LEN * B_DIM)       // 32768
#define KT    32                    // K tiles of 32 (K = 1024)
#define BH    (B_DIM * H_DIM)       // 65536

typedef __attribute__((ext_vector_type(16))) __bf16 v16bf;
typedef __attribute__((ext_vector_type(8)))  float  v8f;

// ---------------- helpers ----------------

__device__ __forceinline__ unsigned short f2bf(float f) {
  unsigned int u = __builtin_bit_cast(unsigned int, f);
  u += 0x7fffu + ((u >> 16) & 1u);   // round-to-nearest-even
  return (unsigned short)(u >> 16);
}

__device__ __forceinline__ float sigf(float x) {
  return 1.0f / (1.0f + __expf(-x));
}
__device__ __forceinline__ float tanh_fast(float x) {
  return 1.0f - 2.0f / (__expf(2.0f * x) + 1.0f);
}

__device__ __forceinline__ v8f wmma_bf16(v16bf a, v16bf b, v8f c) {
  // (neg_a, A, neg_b, B, c_mod, C, reuse_a, reuse_b)
  return __builtin_amdgcn_wmma_f32_16x16x32_bf16(false, a, false, b,
                                                 (short)0, c, false, false);
}

// A-fragment scatter offset for a 16x32 bf16 tile grid over [M,K] with KTiles
// k-tiles. ISA 7.12.2: lanes 0-15 hold K in {0..7,16..23}; lanes 16-31 hold
// {8..15, 24..31}.
__device__ __forceinline__ int apack_off(int m, int k, int KTiles) {
  const int mt = m >> 4, kt = k >> 5;
  const int kk = k & 31;
  const int lane = (m & 15) + (((kk >> 3) & 1) << 4);
  const int e = (kk & 7) + ((kk & 16) ? 8 : 0);
  return (((mt * KTiles + kt) * 32 + lane) << 4) + e;
}

// ---------------- packing kernels ----------------

// input [M_BIG, I] fp32 -> packed bf16 A fragments, tile order [mt][kt].
__global__ void __launch_bounds__(256)
pack_input(const float* __restrict__ src, unsigned short* __restrict__ dst) {
  const int idx  = blockIdx.x * 256 + threadIdx.x;   // < M_BIG*I_DIM
  const int e    = idx & 15;
  const int lane = (idx >> 4) & 31;
  const int tile = idx >> 9;
  const int kt   = tile & (KT - 1);
  const int mt   = tile >> 5;
  const int hi   = lane >> 4;
  const int koff = ((e < 8) ? e : e + 8) + (hi << 3);
  const int m    = (mt << 4) + (lane & 15);
  const int k    = (kt << 5) + koff;
  dst[idx] = f2bf(src[(size_t)m * I_DIM + k]);
}

// W [4096, 1024] fp32 -> packed bf16 B fragments for B[k][n] = W[n][k];
// tile order [nt][kt]. B layout: lane = column; lanes 0-15 hold K in
// {0..7,16..23}; lanes 16-31 hold {8..15, 24..31}.
__global__ void __launch_bounds__(256)
pack_weight(const float* __restrict__ w, unsigned short* __restrict__ dst) {
  const int idx  = blockIdx.x * 256 + threadIdx.x;   // < 4096*1024
  const int e    = idx & 15;
  const int lane = (idx >> 4) & 31;
  const int tile = idx >> 9;
  const int kt   = tile & (KT - 1);
  const int nt   = tile >> 5;
  const int n    = (nt << 4) + (lane & 15);
  const int k    = (kt << 5) + ((e < 8) ? e : e + 8) + ((lane >> 4) << 3);
  dst[idx] = f2bf(w[(size_t)n * 1024 + k]);
}

__global__ void __launch_bounds__(256)
bias_sum(const float* __restrict__ bi, const float* __restrict__ bh,
         float* __restrict__ b) {
  const int n = blockIdx.x * 256 + threadIdx.x;      // < 4096
  b[n] = bi[n] + bh[n];
}

__global__ void __launch_bounds__(256)
init_state(const float* __restrict__ h0, const float* __restrict__ c0,
           float* __restrict__ c, float* __restrict__ h,
           unsigned short* __restrict__ hP) {
  const int idx = blockIdx.x * 256 + threadIdx.x;    // < BH
  const int b = idx >> 10, j = idx & 1023;
  const float hv = h0[idx];
  c[idx] = c0[idx];
  h[idx] = hv;
  hP[apack_off(b, j, KT)] = f2bf(hv);
}

// ---------------- big GEMM: x_gates ----------------
// Wave = 4 Mtiles x 4 Ntiles (64x64 output, 16 wmma / K-step).
// Workgroup = 4 waves spanning 256 output columns. Grid: (M/64, N/256).
__global__ void __launch_bounds__(128)
gemm_xgates(const unsigned short* __restrict__ pA,
            const unsigned short* __restrict__ pB,
            const float* __restrict__ bias,
            float* __restrict__ xg) {
  const int lane = threadIdx.x & 31;
  const int wave = threadIdx.x >> 5;
  const int mt0 = blockIdx.x * 4;
  const int nt0 = blockIdx.y * 16 + wave * 4;

  const unsigned short* aP[4];
  const unsigned short* bP[4];
#pragma unroll
  for (int i = 0; i < 4; ++i)
    aP[i] = pA + (((size_t)(mt0 + i) * KT * 32) + lane) * 16;
#pragma unroll
  for (int j = 0; j < 4; ++j)
    bP[j] = pB + (((size_t)(nt0 + j) * KT * 32) + lane) * 16;

  const v8f zero = {0.f, 0.f, 0.f, 0.f, 0.f, 0.f, 0.f, 0.f};
  v8f acc[4][4];
#pragma unroll
  for (int i = 0; i < 4; ++i)
#pragma unroll
    for (int j = 0; j < 4; ++j) acc[i][j] = zero;

  for (int kt = 0; kt < KT; ++kt) {
    const int off = kt * 512;                 // 32 lanes * 16 elems per tile
    v16bf a[4], b[4];
#pragma unroll
    for (int i = 0; i < 4; ++i) a[i] = *(const v16bf*)(aP[i] + off);
#pragma unroll
    for (int j = 0; j < 4; ++j) b[j] = *(const v16bf*)(bP[j] + off);
    // stream hint (speculative prefetch 8 k-tiles ahead; OOB drops silently)
#pragma unroll
    for (int j = 0; j < 4; ++j)
      __builtin_prefetch(bP[j] + off + 8 * 512, 0, 2);
#pragma unroll
    for (int i = 0; i < 4; ++i)
#pragma unroll
      for (int j = 0; j < 4; ++j)
        acc[i][j] = wmma_bf16(a[i], b[j], acc[i][j]);
  }

  const int hi = lane >> 4;
  const int cl = lane & 15;
#pragma unroll
  for (int j = 0; j < 4; ++j) {
    const int n = (nt0 + j) * 16 + cl;
    const float bv = bias[n];
#pragma unroll
    for (int i = 0; i < 4; ++i) {
#pragma unroll
      for (int r = 0; r < 8; ++r) {
        const int m = (mt0 + i) * 16 + r + hi * 8;
        xg[(size_t)m * G_DIM + n] = acc[i][j][r] + bv;
      }
    }
  }
}

// ---------------- fused recurrent step (K-split + LDS reduce) ----------------
// One kernel per timestep. Workgroup = column tile jt (16 h-columns).
// Each of the 4 waves accumulates a 4(Mtile)x4(gate) WMMA register tile over
// ONE quarter of K (8 k-tiles) -> 4x shorter dependent chain per step.
// Partials are reduced through LDS; wave w then applies the pointwise LSTM +
// retention update for M-tile w (rows w*16..w*16+15) in-register.
// Grid: 64 blocks x 128 threads.
__global__ void __launch_bounds__(128)
lstm_step(const unsigned short* __restrict__ hPi,   // packed h (read)
          const unsigned short* __restrict__ pB,    // packed W_hh
          const float* __restrict__ xg_t,           // x_gates[t] : [64][4096]
          const float* __restrict__ ret,            // retention  : [1024]
          float* __restrict__ c, float* __restrict__ h,
          unsigned short* __restrict__ hPo,         // packed h (write)
          float* __restrict__ out, int t, int last) {
  __shared__ float red[4 * 16 * 32 * 8];            // [kchunk][i*4+s][lane][8] = 64 KB
  const int lane = threadIdx.x & 31;
  const int wave = threadIdx.x >> 5;                // K chunk 0..3
  const int jt = blockIdx.x;                        // column tile 0..63

  const unsigned short* aP[4];
#pragma unroll
  for (int i = 0; i < 4; ++i)
    aP[i] = hPi + (((size_t)i * KT * 32) + lane) * 16;
  const unsigned short* bP[4];
#pragma unroll
  for (int s = 0; s < 4; ++s)                       // gate sections i,f,g,o
    bP[s] = pB + (((size_t)(s * 64 + jt) * KT * 32) + lane) * 16;

  const v8f zero = {0.f, 0.f, 0.f, 0.f, 0.f, 0.f, 0.f, 0.f};
  v8f acc[4][4];                                    // [Mtile][gate]
#pragma unroll
  for (int i = 0; i < 4; ++i)
#pragma unroll
    for (int s = 0; s < 4; ++s) acc[i][s] = zero;

  const int kt0 = wave * (KT / 4);
  for (int kt = kt0; kt < kt0 + KT / 4; ++kt) {
    const int off = kt * 512;
    v16bf a[4], b[4];
#pragma unroll
    for (int i = 0; i < 4; ++i) a[i] = *(const v16bf*)(aP[i] + off);
#pragma unroll
    for (int s = 0; s < 4; ++s) b[s] = *(const v16bf*)(bP[s] + off);
#pragma unroll
    for (int i = 0; i < 4; ++i)
#pragma unroll
      for (int s = 0; s < 4; ++s)
        acc[i][s] = wmma_bf16(a[i], b[s], acc[i][s]);
  }

  // ---- cross-wave K reduction through LDS ----
#pragma unroll
  for (int i = 0; i < 4; ++i)
#pragma unroll
    for (int s = 0; s < 4; ++s)
      *(v8f*)&red[(((wave * 16 + i * 4 + s) * 32) + lane) * 8] = acc[i][s];
  __syncthreads();

  const int i = wave;                               // this wave's M-tile
  v8f sum[4];
#pragma unroll
  for (int s = 0; s < 4; ++s) {
    v8f v = *(const v8f*)&red[(((0 * 16 + i * 4 + s) * 32) + lane) * 8];
#pragma unroll
    for (int p = 1; p < 4; ++p)
      v = v + *(const v8f*)&red[(((p * 16 + i * 4 + s) * 32) + lane) * 8];
    sum[s] = v;
  }

  // ---- fused pointwise LSTM + retention, in-register ----
  const int hi = lane >> 4;
  const int cl = lane & 15;
  const int j = jt * 16 + cl;                       // h column, 0..1023
  const float rr = ret[j];
  const float* xgc = xg_t + j;
#pragma unroll
  for (int r = 0; r < 8; ++r) {
    const int b = i * 16 + r + hi * 8;              // batch row, 0..63
    const size_t grow = (size_t)b * G_DIM;
    const float ig = sigf     (sum[0][r] + xgc[grow]);
    const float fg = sigf     (sum[1][r] + xgc[grow + 1024]);
    const float gg = tanh_fast(sum[2][r] + xgc[grow + 2048]);
    const float og = sigf     (sum[3][r] + xgc[grow + 3072]);
    const int idx = b * H_DIM + j;
    const float cy  = fg * c[idx] + ig * gg;
    const float hy0 = og * tanh_fast(cy);
    const float hy  = rr * h[idx] + (1.0f - rr) * hy0;
    c[idx] = cy;
    h[idx] = hy;
    out[(size_t)t * BH + idx] = hy;
    hPo[apack_off(b, j, KT)] = f2bf(hy);
    if (last) {
      out[(size_t)S_LEN * BH + idx] = hy;           // final h
      out[(size_t)S_LEN * BH + BH + idx] = cy;      // final c
    }
  }
}

// ---------------- host ----------------

extern "C" void kernel_launch(void* const* d_in, const int* in_sizes, int n_in,
                              void* d_out, int out_size, void* d_ws, size_t ws_size,
                              hipStream_t stream) {
  const float* input_    = (const float*)d_in[0];   // [S,B,I]
  const float* h0        = (const float*)d_in[1];   // [B,H]
  const float* c0        = (const float*)d_in[2];   // [B,H]
  const float* weight_ih = (const float*)d_in[3];   // [4H,I]
  const float* weight_hh = (const float*)d_in[4];   // [4H,H]
  const float* bias_ih   = (const float*)d_in[5];   // [4H]
  const float* bias_hh   = (const float*)d_in[6];   // [4H]
  const float* ret       = (const float*)d_in[7];   // [H]
  float* out = (float*)d_out;

  char* ws = (char*)d_ws;
  // workspace layout (all offsets 32B+ aligned)
  unsigned short* pA_in = (unsigned short*)(ws);                      // 64 MB
  unsigned short* pB_ih = (unsigned short*)(ws + 67108864ull);        //  8 MB
  unsigned short* pB_hh = (unsigned short*)(ws + 75497472ull);        //  8 MB
  float*          bias  = (float*)(ws + 83886080ull);                 // 16 KB
  float*          xg    = (float*)(ws + 83902464ull);                 // 512 MB
  float*          cbuf  = (float*)(ws + 620773376ull);                // 256 KB
  float*          hbuf  = (float*)(ws + 621035520ull);                // 256 KB
  unsigned short* hP0   = (unsigned short*)(ws + 621297664ull);       // 128 KB
  unsigned short* hP1   = (unsigned short*)(ws + 621428736ull);       // 128 KB

  // 1) pack / convert
  pack_input <<<(M_BIG * I_DIM) / 256, 256, 0, stream>>>(input_, pA_in);
  pack_weight<<<(G_DIM * I_DIM) / 256, 256, 0, stream>>>(weight_ih, pB_ih);
  pack_weight<<<(G_DIM * H_DIM) / 256, 256, 0, stream>>>(weight_hh, pB_hh);
  bias_sum   <<<G_DIM / 256, 256, 0, stream>>>(bias_ih, bias_hh, bias);
  init_state <<<BH / 256, 256, 0, stream>>>(h0, c0, cbuf, hbuf, hP0);

  // 2) input projection for all timesteps (big WMMA GEMM)
  gemm_xgates<<<dim3(M_BIG / 64, G_DIM / 256), 128, 0, stream>>>(
      pA_in, pB_ih, bias, xg);

  // 3) sequential recurrence: one fused kernel per step, ping-pong packed h
  for (int t = 0; t < S_LEN; ++t) {
    const float* xg_t = xg + (size_t)t * B_DIM * G_DIM;
    unsigned short* hin  = (t & 1) ? hP1 : hP0;
    unsigned short* hout = (t & 1) ? hP0 : hP1;
    lstm_step<<<64, 128, 0, stream>>>(hin, pB_hh, xg_t, ret, cbuf, hbuf,
                                      hout, out, t, (t == S_LEN - 1) ? 1 : 0);
  }
}
